// PatchKoopmanODE_47605417509227
// MI455X (gfx1250) — matrix operand order repaired and checked by reading, not compiled, not run on hardware
//
#include <hip/hip_runtime.h>
#include <hip/hip_bf16.h>

typedef _Float16 half_t;
typedef _Float16 v16h __attribute__((ext_vector_type(16)));
typedef float    v8f  __attribute__((ext_vector_type(8)));

#define B_SZ 2048
#define SEQ 512
#define PRED 336
#define PLEN 16
#define STRIDE_ 8
#define CH 16
#define DM 128
#define HID 256
#define PR 64
#define NP 63
#define TP 42
#define KCMP (DM * NP) /* 8064 */

__device__ __forceinline__ float silu_f(float x) { return x / (1.0f + __expf(-x)); }

// A-matrix fragment (16x32 f16): row = lane&15; lane<16 holds K {0..7,16..23}, lane>=16 holds K {8..15,24..31}
__device__ __forceinline__ v16h frag_a(const half_t* base, int stride, int k0, int lane) {
  v16h a;
  int row = lane & 15;
  int h8 = (lane & 16) ? 8 : 0;
#pragma unroll
  for (int i = 0; i < 16; ++i) {
    int v = i >> 1, e = i & 1;
    int k = k0 + ((v >> 2) << 4) + h8 + ((v & 3) << 1) + e;
    a[i] = base[row * stride + k];
  }
  return a;
}

// B-matrix fragment (32x16 f16), B[k,n] = W[n,k] (W row-major n x K):
// n = lane&15; lane<16 holds K {0..7,16..23}, lane>=16 holds K {8..15,24..31}
__device__ __forceinline__ v16h frag_b(const half_t* W, int wstride, int nbase, int k0, int lane) {
  v16h b;
  int n = nbase + (lane & 15);
  const half_t* p = W + (size_t)n * wstride + k0 + ((lane & 16) ? 8 : 0);
#pragma unroll
  for (int i = 0; i < 16; ++i) b[i] = p[(i & 7) + ((i >> 3) << 4)];
  return b;
}

#define WMMA_F16(a, b, c) \
  __builtin_amdgcn_wmma_f32_16x16x32_f16(false, (a), false, (b), (short)0, (c), false, false)

// ---------------- weight f32 -> f16 ----------------
__global__ void k_h(const float* __restrict__ s, half_t* __restrict__ d, int n) {
  for (int i = blockIdx.x * blockDim.x + threadIdx.x; i < n; i += gridDim.x * blockDim.x)
    d[i] = (half_t)s[i];
}

// ---------------- patch embed: pe[b,p,:] = patch(b,p) @ W_pe^T + b_pe ----------------
__global__ void k_patch_embed(const float* __restrict__ x, const half_t* __restrict__ Wpe,
                              const float* __restrict__ bpe, float* __restrict__ pe) {
  __shared__ half_t sA[16 * 256];
  int b = blockIdx.x, pbase = blockIdx.y * 16;
  int tid = threadIdx.x, lane = tid & 31, wave = tid >> 5;
  // stage 16 patches (gather directly from x; feature k = c*16 + t)
  for (int idx = tid; idx < 16 * 256; idx += 128) {
    int r = idx >> 8, k = idx & 255;
    int p = pbase + r, c = k >> 4, t = k & 15;
    float v = 0.f;
    if (p < NP) v = x[((size_t)b * SEQ + p * STRIDE_ + t) * CH + c];
    sA[idx] = (half_t)v;
  }
  __syncthreads();
#pragma unroll
  for (int j = 0; j < 2; ++j) {
    int nt = wave * 2 + j;
    v8f acc = {};
#pragma unroll
    for (int kc = 0; kc < 8; ++kc) {
      v16h a  = frag_a(sA, 256, kc * 32, lane);
      v16h bb = frag_b(Wpe, 256, nt * 16, kc * 32, lane);
      acc = WMMA_F16(a, bb, acc);
    }
    int col = nt * 16 + (lane & 15);
    int h8 = (lane & 16) ? 8 : 0;
#pragma unroll
    for (int v = 0; v < 8; ++v) {
      int p = pbase + v + h8;
      if (p < NP) pe[((size_t)b * NP + p) * DM + col] = acc[v] + bpe[col];
    }
  }
}

// ---------------- depthwise conv (k=3, pad 2, keep first 63) + silu -> patch_flat f16 ----------------
__global__ void k_conv(const float* __restrict__ pe, const float* __restrict__ Wc,
                       const float* __restrict__ bc, half_t* __restrict__ pf) {
  size_t N = (size_t)B_SZ * NP * DM;
  for (size_t idx = blockIdx.x * (size_t)blockDim.x + threadIdx.x; idx < N;
       idx += (size_t)gridDim.x * blockDim.x) {
    int d = (int)(idx % DM);
    size_t r = idx / DM;
    int i = (int)(r % NP);
    size_t b = r / NP;
    float v = 0.f;
#pragma unroll
    for (int kk = 0; kk < 3; ++kk) {
      int ii = i + kk - 2;
      if (ii >= 0 && ii < NP) v += pe[(b * NP + ii) * DM + d] * Wc[d * 3 + kk];
    }
    pf[b * KCMP + (size_t)i * DM + d] = (half_t)silu_f(v + bc[d]);
  }
}

// ---------------- z0 = pe[:, 62, :] ----------------
__global__ void k_zinit(const float* __restrict__ pe, float* __restrict__ z0) {
  int idx = blockIdx.x * blockDim.x + threadIdx.x;
  if (idx < B_SZ * DM) {
    int b = idx >> 7, d = idx & 127;
    z0[idx] = pe[((size_t)b * NP + (NP - 1)) * DM + d];
  }
}

// ---------------- a0 = patch_flat @ W_cmp^T + b_cmp  (M=2048, K=8064, N=64) ----------------
__global__ void k_cmp(const half_t* __restrict__ pf, const half_t* __restrict__ Wcmp,
                      const float* __restrict__ bcmp, float* __restrict__ a0) {
  __shared__ half_t sA[16 * 32];
  int mbase = blockIdx.x * 16;
  int tid = threadIdx.x, lane = tid & 31, wave = tid >> 5;
  v8f acc = {};
  for (int kc = 0; kc < KCMP / 32; ++kc) {
    for (int idx = tid; idx < 512; idx += 128) {
      int r = idx >> 5, kk = idx & 31;
      sA[idx] = pf[(size_t)(mbase + r) * KCMP + kc * 32 + kk];
    }
    if (kc + 1 < KCMP / 32)  // gfx1250 global_prefetch of next A chunk
      __builtin_prefetch(pf + (size_t)(mbase + (tid & 15)) * KCMP + (kc + 1) * 32, 0, 1);
    __syncthreads();
    v16h a  = frag_a(sA, 32, 0, lane);
    v16h bb = frag_b(Wcmp, KCMP, wave * 16, kc * 32, lane);
    acc = WMMA_F16(a, bb, acc);
    __syncthreads();
  }
  int col = wave * 16 + (lane & 15);
  int h8 = (lane & 16) ? 8 : 0;
#pragma unroll
  for (int v = 0; v < 8; ++v)
    a0[(size_t)(mbase + v + h8) * PR + col] = acc[v] + bcmp[col];
}

// ---------------- 42-step Euler scan: each block owns 16 batch rows for all steps ----------------
__global__ void k_scan(const float* __restrict__ a0, const float* __restrict__ z0,
                       const half_t* __restrict__ Wn1h, const float* __restrict__ bn1,
                       const half_t* __restrict__ Wn2h, const float* __restrict__ bn2,
                       const half_t* __restrict__ Wl1h, const float* __restrict__ bl1,
                       const half_t* __restrict__ Wl2h, const float* __restrict__ bl2,
                       const float* __restrict__ spanA, half_t* __restrict__ ztraj) {
  __shared__ half_t sAh[16 * 64];
  __shared__ half_t sH[16 * 512];   // [h1 | h2]
  __shared__ float  sa[16 * 64];
  __shared__ float  sz[16 * 128];
  __shared__ float  sc[16 * 128];
  __shared__ float  sda[16 * 64];
  __shared__ float  sdz[16 * 128];
  int bbase = blockIdx.x * 16;
  int tid = threadIdx.x, lane = tid & 31, wave = tid >> 5;
  float dt = fminf(fmaxf(spanA[0], 1e-8f), 5.0f);
  for (int idx = tid; idx < 16 * 64; idx += 128)
    sa[idx] = a0[(size_t)(bbase + (idx >> 6)) * PR + (idx & 63)];
  for (int idx = tid; idx < 16 * 128; idx += 128)
    sz[idx] = z0[(size_t)(bbase + (idx >> 7)) * DM + (idx & 127)];
  __syncthreads();

  for (int t = 0; t < TP; ++t) {
    for (int idx = tid; idx < 16 * 64; idx += 128) sAh[idx] = (half_t)sa[idx];
    __syncthreads();
    // stage 1: h = silu(a @ [Wn1;Wl1]^T + bias)  -> 16 x 512
#pragma unroll
    for (int j = 0; j < 8; ++j) {
      int nt = wave + 4 * j;  // uniform per wave
      v8f acc = {};
#pragma unroll
      for (int kc = 0; kc < 2; ++kc) {
        v16h a  = frag_a(sAh, 64, kc * 32, lane);
        v16h bb = (nt < 16) ? frag_b(Wn1h, 64, nt * 16, kc * 32, lane)
                            : frag_b(Wl1h, 64, (nt - 16) * 16, kc * 32, lane);
        acc = WMMA_F16(a, bb, acc);
      }
      int n = nt * 16 + (lane & 15);
      int h8 = (lane & 16) ? 8 : 0;
      float bias = (nt < 16) ? bn1[n] : bl1[n - 256];
#pragma unroll
      for (int v = 0; v < 8; ++v) sH[(v + h8) * 512 + n] = (half_t)silu_f(acc[v] + bias);
    }
    __syncthreads();
    // stage 2: da = tanh(h1 @ Wn2^T + bn2) (16x64); c = tanh(h2 @ Wl2^T + bl2) (16x128)
#pragma unroll
    for (int j = 0; j < 3; ++j) {
      int nt = wave * 3 + j;          // 0..11, uniform per wave
      bool is_da = (nt < 4);
      int ntc = is_da ? nt : nt - 4;
      int abase = is_da ? 0 : 256;
      const half_t* W = is_da ? Wn2h : Wl2h;
      v8f acc = {};
#pragma unroll
      for (int kc = 0; kc < 8; ++kc) {
        v16h a  = frag_a(sH, 512, abase + kc * 32, lane);
        v16h bb = frag_b(W, 256, ntc * 16, kc * 32, lane);
        acc = WMMA_F16(a, bb, acc);
      }
      int col = ntc * 16 + (lane & 15);
      int h8 = (lane & 16) ? 8 : 0;
#pragma unroll
      for (int v = 0; v < 8; ++v) {
        int row = v + h8;
        if (is_da) sda[row * 64 + col]  = tanhf(acc[v] + bn2[col]);
        else       sc[row * 128 + col]  = tanhf(acc[v] + bl2[col]);
      }
    }
    __syncthreads();
    // dz[row,i] = sum_k c[row,k] * z[row,(i+k)%128]  (circular correlation)
    // Register-blocked sliding window: per 16-wide k block, 48 LDS reads feed 256 FMAs.
    {
      int row = tid >> 3, i0 = (tid & 7) * 16;
      const float* crow = &sc[row * 128];
      const float* zrow = &sz[row * 128];
      float s[16];
#pragma unroll
      for (int q = 0; q < 16; ++q) s[q] = 0.f;
      for (int kb = 0; kb < 8; ++kb) {
        int base = (i0 + kb * 16) & 127;
        float w[32];
#pragma unroll
        for (int m = 0; m < 32; ++m) {
          int idx = base + m;
          w[m] = zrow[idx & 127];
        }
        float cc[16];
#pragma unroll
        for (int kk = 0; kk < 16; ++kk) cc[kk] = crow[kb * 16 + kk];
#pragma unroll
        for (int kk = 0; kk < 16; ++kk)
#pragma unroll
          for (int q = 0; q < 16; ++q) s[q] += cc[kk] * w[kk + q];
      }
#pragma unroll
      for (int q = 0; q < 16; ++q) sdz[row * 128 + i0 + q] = s[q];
    }
    __syncthreads();
    for (int idx = tid; idx < 16 * 64; idx += 128) sa[idx] += dt * sda[idx];
    for (int idx = tid; idx < 16 * 128; idx += 128) {
      float zn = sz[idx] + dt * sdz[idx];
      sz[idx] = zn;
      ztraj[((size_t)t * B_SZ + bbase + (idx >> 7)) * DM + (idx & 127)] = (half_t)zn;
    }
    __syncthreads();
  }
}

// ---------------- zero the overlap-add accumulator ----------------
__global__ void k_zero(float* __restrict__ p, size_t n) {
  for (size_t i = blockIdx.x * (size_t)blockDim.x + threadIdx.x; i < n;
       i += (size_t)gridDim.x * blockDim.x)
    p[i] = 0.f;
}

// ---------------- decoder: dec_t = silu(z_t @ Wd1^T + bd1) @ Wd2^T + bd2 ; scatter-add ----------------
__global__ void k_dec(const half_t* __restrict__ ztraj,
                      const half_t* __restrict__ Wd1h, const float* __restrict__ bd1,
                      const half_t* __restrict__ Wd2h, const float* __restrict__ bd2,
                      float* __restrict__ accbuf) {
  __shared__ half_t sZ[16 * 128];
  __shared__ half_t sH[16 * 256];
  int t = blockIdx.y, bbase = blockIdx.x * 16;
  int tid = threadIdx.x, lane = tid & 31, wave = tid >> 5;
  for (int idx = tid; idx < 16 * 128; idx += 128)
    sZ[idx] = ztraj[((size_t)t * B_SZ + bbase + (idx >> 7)) * DM + (idx & 127)];
  __syncthreads();
#pragma unroll
  for (int j = 0; j < 4; ++j) {
    int nt = wave + 4 * j;
    v8f acc = {};
#pragma unroll
    for (int kc = 0; kc < 4; ++kc) {
      v16h a  = frag_a(sZ, 128, kc * 32, lane);
      v16h bb = frag_b(Wd1h, 128, nt * 16, kc * 32, lane);
      acc = WMMA_F16(a, bb, acc);
    }
    int n = nt * 16 + (lane & 15);
    int h8 = (lane & 16) ? 8 : 0;
#pragma unroll
    for (int v = 0; v < 8; ++v) sH[(v + h8) * 256 + n] = (half_t)silu_f(acc[v] + bd1[n]);
  }
  __syncthreads();
#pragma unroll
  for (int j = 0; j < 4; ++j) {
    int nt = wave + 4 * j;
    v8f acc = {};
#pragma unroll
    for (int kc = 0; kc < 8; ++kc) {
      v16h a  = frag_a(sH, 256, kc * 32, lane);
      v16h bb = frag_b(Wd2h, 256, nt * 16, kc * 32, lane);
      acc = WMMA_F16(a, bb, acc);
    }
    int n = nt * 16 + (lane & 15);
    int h8 = (lane & 16) ? 8 : 0;
    int k16 = n >> 4, c = n & 15;
    int p = t * STRIDE_ + k16;
    if (p < PRED) {
#pragma unroll
      for (int v = 0; v < 8; ++v)
        atomicAdd(&accbuf[((size_t)(bbase + v + h8) * PRED + p) * CH + c], acc[v] + bd2[n]);
    }
  }
}

// ---------------- finalize: out = [x, acc/(count+1e-6)] ----------------
__global__ void k_final(const float* __restrict__ x, const float* __restrict__ accbuf,
                        float* __restrict__ out) {
  size_t N1 = (size_t)B_SZ * SEQ * CH;
  size_t N2 = (size_t)B_SZ * PRED * CH;
  for (size_t idx = blockIdx.x * (size_t)blockDim.x + threadIdx.x; idx < N1 + N2;
       idx += (size_t)gridDim.x * blockDim.x) {
    if (idx < N1) {
      out[idx] = x[idx];
    } else {
      size_t i2 = idx - N1;
      int p = (int)((i2 / CH) % PRED);
      float cnt = (p < STRIDE_) ? 1.0f : 2.0f;
      out[idx] = accbuf[i2] / (cnt + 1e-6f);
    }
  }
}

extern "C" void kernel_launch(void* const* d_in, const int* in_sizes, int n_in,
                              void* d_out, int out_size, void* d_ws, size_t ws_size,
                              hipStream_t stream) {
  (void)in_sizes; (void)n_in; (void)out_size; (void)ws_size;
  const float* x     = (const float*)d_in[0];
  const float* Wpe   = (const float*)d_in[1];
  const float* bpe   = (const float*)d_in[2];
  const float* Wconv = (const float*)d_in[3];
  const float* bconv = (const float*)d_in[4];
  const float* Wcmp  = (const float*)d_in[5];
  const float* bcmp  = (const float*)d_in[6];
  const float* Wn1   = (const float*)d_in[7];
  const float* bn1   = (const float*)d_in[8];
  const float* Wn2   = (const float*)d_in[9];
  const float* bn2   = (const float*)d_in[10];
  const float* Wl1   = (const float*)d_in[11];
  const float* bl1   = (const float*)d_in[12];
  const float* Wl2   = (const float*)d_in[13];
  const float* bl2   = (const float*)d_in[14];
  const float* spanA = (const float*)d_in[15];
  const float* Wd1   = (const float*)d_in[16];
  const float* bd1   = (const float*)d_in[17];
  const float* Wd2   = (const float*)d_in[18];
  const float* bd2   = (const float*)d_in[19];
  float* out = (float*)d_out;

  char* w = (char*)d_ws;
  size_t off = 0;
  auto take = [&](size_t bytes) -> void* {
    void* p = w + off;
    off = (off + bytes + 255) & ~(size_t)255;
    return p;
  };
  half_t* Wpe_h  = (half_t*)take(sizeof(half_t) * DM * 256);
  half_t* Wcmp_h = (half_t*)take(sizeof(half_t) * PR * KCMP);
  half_t* Wn1_h  = (half_t*)take(sizeof(half_t) * HID * PR);
  half_t* Wn2_h  = (half_t*)take(sizeof(half_t) * PR * HID);
  half_t* Wl1_h  = (half_t*)take(sizeof(half_t) * HID * PR);
  half_t* Wl2_h  = (half_t*)take(sizeof(half_t) * DM * HID);
  half_t* Wd1_h  = (half_t*)take(sizeof(half_t) * 256 * DM);
  half_t* Wd2_h  = (half_t*)take(sizeof(half_t) * 256 * 256);
  float*  pe     = (float*)take(sizeof(float) * (size_t)B_SZ * NP * DM);
  half_t* pf     = (half_t*)take(sizeof(half_t) * (size_t)B_SZ * KCMP);
  float*  a0     = (float*)take(sizeof(float) * (size_t)B_SZ * PR);
  float*  z0     = (float*)take(sizeof(float) * (size_t)B_SZ * DM);
  half_t* ztraj  = (half_t*)take(sizeof(half_t) * (size_t)TP * B_SZ * DM);
  float*  accb   = (float*)take(sizeof(float) * (size_t)B_SZ * PRED * CH);

  // weight conversions (f32 -> f16), once per launch
  k_h<<<64, 256, 0, stream>>>(Wpe,  Wpe_h,  DM * 256);
  k_h<<<512, 256, 0, stream>>>(Wcmp, Wcmp_h, PR * KCMP);
  k_h<<<64, 256, 0, stream>>>(Wn1,  Wn1_h,  HID * PR);
  k_h<<<64, 256, 0, stream>>>(Wn2,  Wn2_h,  PR * HID);
  k_h<<<64, 256, 0, stream>>>(Wl1,  Wl1_h,  HID * PR);
  k_h<<<64, 256, 0, stream>>>(Wl2,  Wl2_h,  DM * HID);
  k_h<<<64, 256, 0, stream>>>(Wd1,  Wd1_h,  256 * DM);
  k_h<<<128, 256, 0, stream>>>(Wd2,  Wd2_h,  256 * 256);

  k_patch_embed<<<dim3(B_SZ, 4), 128, 0, stream>>>(x, Wpe_h, bpe, pe);
  k_conv<<<4096, 256, 0, stream>>>(pe, Wconv, bconv, pf);
  k_zinit<<<(B_SZ * DM + 255) / 256, 256, 0, stream>>>(pe, z0);
  k_cmp<<<B_SZ / 16, 128, 0, stream>>>(pf, Wcmp_h, bcmp, a0);
  k_scan<<<B_SZ / 16, 128, 0, stream>>>(a0, z0, Wn1_h, bn1, Wn2_h, bn2,
                                        Wl1_h, bl1, Wl2_h, bl2, spanA, ztraj);
  k_zero<<<4096, 256, 0, stream>>>(accb, (size_t)B_SZ * PRED * CH);
  k_dec<<<dim3(B_SZ / 16, TP), 128, 0, stream>>>(ztraj, Wd1_h, bd1, Wd2_h, bd2, accb);
  k_final<<<8192, 256, 0, stream>>>(x, accb, out);
}
